// GPSDiffusion2_Conditional_77206332113088
// MI455X (gfx1250) — compile-verified, matrix operand order checked by reading, not run on hardware
//
#include <hip/hip_runtime.h>
#include <hip/hip_bf16.h>

typedef __attribute__((ext_vector_type(16))) _Float16 v16h;
typedef __attribute__((ext_vector_type(8)))  _Float16 v8h;
typedef __attribute__((ext_vector_type(8)))  float    v8f;

#define NB   512
#define HID  128
#define DIN  4096
#define HEADS 4
#define DH   32
#define LLY  5

// ---------------------------------------------------------------------------
// WMMA fragment load: row-major [rows, ld] f16 source; 16x32 tile at (row0,k0).
// 16-bit A-matrix layout: lanes 0-15 -> M=row, K = k0+[0..7], k0+[16..23];
// lanes 16-31 -> same rows, K = k0+[8..15], k0+[24..31].
// W (= B^T) uses the identical scheme, so y = x @ W.T needs no transpose.
// ---------------------------------------------------------------------------
__device__ __forceinline__ v16h load_frag16(const _Float16* __restrict__ base,
                                            int ld, int row0, int k0) {
  int lane = threadIdx.x & 31;
  int r    = lane & 15;
  int hi   = (lane >> 4) & 1;
  const _Float16* p = base + (size_t)(row0 + r) * (size_t)ld + k0 + hi * 8;
  v8h lo = *(const v8h*)(p);
  v8h up = *(const v8h*)(p + 16);
  v16h out;
#pragma unroll
  for (int i = 0; i < 8; ++i) { out[i] = lo[i]; out[i + 8] = up[i]; }
  return out;
}

__device__ __forceinline__ void store_sub16(float* __restrict__ Cf, int ldcf,
                                            _Float16* __restrict__ Ch, int ldch,
                                            const float* __restrict__ bias,
                                            v8f acc, int m0, int n0,
                                            float alpha, int relu) {
  int lane  = threadIdx.x & 31;
  int col   = n0 + (lane & 15);
  int rbase = m0 + ((lane >> 4) << 3);
  float bv = bias ? bias[col] : 0.0f;
#pragma unroll
  for (int i = 0; i < 8; ++i) {
    float v = acc[i] * alpha + bv;
    if (relu) v = v > 0.f ? v : 0.f;
    int row = rbase + i;
    if (Cf) Cf[(size_t)row * ldcf + col] = v;
    if (Ch) Ch[(size_t)row * ldch + col] = (_Float16)v;
  }
}

// ---------------------------------------------------------------------------
// Generic GEMM: C[M,N] = alpha * (A[M,K] @ W[N,K]^T) + bias, optional ReLU,
// optional f32 and/or f16 outputs. One wave per 32x32 C macro-tile
// (2x2 WMMA sub-tiles -> 4 WMMAs per k-step from 8 b128 loads), 4 waves/block.
// M and N must be multiples of 32, K a multiple of 32 (true for all calls).
// ---------------------------------------------------------------------------
__global__ void __launch_bounds__(128)
gemm_wmma_f16(const _Float16* __restrict__ A, int lda,
              const _Float16* __restrict__ W, int ldw,
              const float* __restrict__ bias,
              float* __restrict__ Cf, int ldcf,
              _Float16* __restrict__ Ch, int ldch,
              int M, int N, int K, float alpha, int relu) {
  int wave = (int)(threadIdx.x >> 5);
  int tiles_n = N >> 5;
  int total   = (M >> 5) * tiles_n;
  int tile    = blockIdx.x * 4 + wave;
  if (tile >= total) return;                    // wave-uniform; EXEC stays full
  int tm = tile / tiles_n, tn = tile - tm * tiles_n;
  int m0 = tm << 5, n0 = tn << 5;
  v8f acc00 = {}, acc01 = {}, acc10 = {}, acc11 = {};
  for (int k0 = 0; k0 < K; k0 += 32) {
    v16h a0 = load_frag16(A, lda, m0,      k0);
    v16h a1 = load_frag16(A, lda, m0 + 16, k0);
    v16h b0 = load_frag16(W, ldw, n0,      k0);
    v16h b1 = load_frag16(W, ldw, n0 + 16, k0);
    acc00 = __builtin_amdgcn_wmma_f32_16x16x32_f16(false, a0, false, b0,
                                                   (short)0, acc00, false, false);
    acc01 = __builtin_amdgcn_wmma_f32_16x16x32_f16(false, a0, false, b1,
                                                   (short)0, acc01, false, false);
    acc10 = __builtin_amdgcn_wmma_f32_16x16x32_f16(false, a1, false, b0,
                                                   (short)0, acc10, false, false);
    acc11 = __builtin_amdgcn_wmma_f32_16x16x32_f16(false, a1, false, b1,
                                                   (short)0, acc11, false, false);
  }
  store_sub16(Cf, ldcf, Ch, ldch, bias, acc00, m0,      n0,      alpha, relu);
  store_sub16(Cf, ldcf, Ch, ldch, bias, acc01, m0,      n0 + 16, alpha, relu);
  store_sub16(Cf, ldcf, Ch, ldch, bias, acc10, m0 + 16, n0,      alpha, relu);
  store_sub16(Cf, ldcf, Ch, ldch, bias, acc11, m0 + 16, n0 + 16, alpha, relu);
}

// --------------------------- small helper kernels ---------------------------

__global__ void f32_to_f16_kernel(const float* __restrict__ s,
                                  _Float16* __restrict__ d, int n) {
  int i = blockIdx.x * 256 + threadIdx.x;
  if (i < n) d[i] = (_Float16)s[i];
}

// cat_w[:, :128] strided f16 copy (src ld = 144)
__global__ void catw_conv_kernel(const float* __restrict__ s,
                                 _Float16* __restrict__ d) {
  int i = blockIdx.x * 256 + threadIdx.x;
  if (i < HID * HID) { int r = i >> 7, c = i & 127; d[i] = (_Float16)s[r * 144 + c]; }
}

// Time embedding + label embedding -> hcat[:,128:384] (f16)
__global__ void __launch_bounds__(128)
embed_kernel(const float* __restrict__ t, const int* __restrict__ y,
             const float* __restrict__ fr,
             const float* __restrict__ w1, const float* __restrict__ b1,
             const float* __restrict__ w2, const float* __restrict__ b2,
             const float* __restrict__ tw, const float* __restrict__ tb,
             const float* __restrict__ lemb, _Float16* __restrict__ hcat) {
  __shared__ float e0[16], e1[64], e2[64];
  int n = blockIdx.x, tid = threadIdx.x;
  float tv = t[n];
  if (tid < 16) {
    float tp = tv * fr[tid & 7] * 6.28318530717958647692f;
    e0[tid] = (tid < 8) ? sinf(tp) : cosf(tp);
  }
  __syncthreads();
  if (tid < 64) {
    float s = b1[tid];
    for (int j = 0; j < 16; ++j) s += w1[tid * 16 + j] * e0[j];
    e1[tid] = s / (1.f + __expf(-s));           // SiLU
  }
  __syncthreads();
  if (tid < 64) {
    float s = b2[tid];
    for (int j = 0; j < 64; ++j) s += w2[tid * 64 + j] * e1[j];
    e2[tid] = s;
  }
  __syncthreads();
  float s = tb[tid];
  for (int j = 0; j < 64; ++j) s += tw[tid * 64 + j] * e2[j];
  hcat[(size_t)n * 384 + 128 + tid] = (_Float16)s;
  hcat[(size_t)n * 384 + 256 + tid] = (_Float16)lemb[y[n] * HID + tid];
}

// Constant PE contribution: fully-connected graph => pe row == pen_b exactly.
__global__ void catbias_kernel(const float* __restrict__ pen_b,
                               const float* __restrict__ pel_w,
                               const float* __restrict__ pel_b,
                               const float* __restrict__ cat_w,
                               const float* __restrict__ cat_b,
                               float* __restrict__ out) {
  __shared__ float pe[16];
  int tid = threadIdx.x;
  if (tid < 16) {
    float s = pel_b[tid];
    for (int k = 0; k < 20; ++k) s += pel_w[tid * 20 + k] * pen_b[k];
    pe[tid] = s;
  }
  __syncthreads();
  float s = cat_b[tid];
  for (int j = 0; j < 16; ++j) s += cat_w[tid * 144 + 128 + j] * pe[j];
  out[tid] = s;
}

// Graph aggregation collapses to a single vector on the complete graph:
// gi = (colsum(h) @ gg_w) @ gru_wih^T + gru_bih
__global__ void __launch_bounds__(384)
vec_kernel(const float* __restrict__ h, const float* __restrict__ gg_w,
           const float* __restrict__ wih, const float* __restrict__ bih,
           float* __restrict__ gi) {
  __shared__ float hs[HID], ag[HID];
  int tid = threadIdx.x;
  if (tid < HID) {
    float s = 0.f;
    for (int n = 0; n < NB; ++n) s += h[n * HID + tid];
    hs[tid] = s;
  }
  __syncthreads();
  if (tid < HID) {
    float s = 0.f;
    for (int k = 0; k < HID; ++k) s += hs[k] * gg_w[k * HID + tid];
    ag[tid] = s;
  }
  __syncthreads();
  float s = bih[tid];
  for (int j = 0; j < HID; ++j) s += wih[tid * HID + j] * ag[j];
  gi[tid] = s;
}

// GRU gate combine + residual: p = ((1-z)*n + z*h) + h
__global__ void gru_elt_kernel(const float* __restrict__ GH,
                               const float* __restrict__ gi,
                               const float* __restrict__ h,
                               float* __restrict__ p) {
  int i = blockIdx.x * 256 + threadIdx.x;
  if (i >= NB * HID) return;
  int n = i >> 7, f = i & 127;
  float r  = 1.f / (1.f + __expf(-(gi[f]        + GH[n * 384 + f])));
  float z  = 1.f / (1.f + __expf(-(gi[128 + f]  + GH[n * 384 + 128 + f])));
  float nn = tanhf(gi[256 + f] + r * GH[n * 384 + 256 + f]);
  float hv = h[i];
  p[i] = (1.f - z) * nn + z * hv + hv;
}

// BatchNorm over the batch dim (biased var), one feature per block.
__global__ void __launch_bounds__(256)
bn_kernel(const float* __restrict__ src, const float* __restrict__ g,
          const float* __restrict__ b, float* __restrict__ of,
          _Float16* __restrict__ oh) {
  __shared__ float r1[256], r2[256];
  int f = blockIdx.x, tid = threadIdx.x;
  float s = 0.f, s2 = 0.f;
  for (int n = tid; n < NB; n += 256) {
    float v = src[n * HID + f]; s += v; s2 += v * v;
  }
  r1[tid] = s; r2[tid] = s2; __syncthreads();
  for (int o = 128; o > 0; o >>= 1) {
    if (tid < o) { r1[tid] += r1[tid + o]; r2[tid] += r2[tid + o]; }
    __syncthreads();
  }
  float mean = r1[0] * (1.f / NB);
  float var  = r2[0] * (1.f / NB) - mean * mean;
  float inv  = rsqrtf(var + 1e-5f) * g[f];
  float bb   = b[f];
  for (int n = tid; n < NB; n += 256) {
    float v = (src[n * HID + f] - mean) * inv + bb;
    of[n * HID + f] = v;
    if (oh) oh[n * HID + f] = (_Float16)v;
  }
}

__global__ void add_kernel(const float* __restrict__ a, const float* __restrict__ b,
                           float* __restrict__ of, _Float16* __restrict__ oh, int n) {
  int i = blockIdx.x * 256 + threadIdx.x;
  if (i < n) { float v = a[i] + b[i]; of[i] = v; if (oh) oh[i] = (_Float16)v; }
}

// Build per-head transposed V (W-form for the o = softmax @ v GEMM): [4][32][512]
__global__ void vt_kernel(const float* __restrict__ qkv, _Float16* __restrict__ vt) {
  int i = blockIdx.x * 256 + threadIdx.x;
  if (i >= HEADS * DH * NB) return;
  int n = i & 511, d = (i >> 9) & 31, h = i >> 14;
  vt[i] = (_Float16)qkv[(size_t)n * 384 + 256 + h * DH + d];
}

// Row softmax over S[2048][512] (head-major), f16 output.
__global__ void __launch_bounds__(256)
softmax_kernel(const float* __restrict__ S, _Float16* __restrict__ A) {
  __shared__ float red[256];
  int row = blockIdx.x, tid = threadIdx.x;
  const float* s = S + (size_t)row * NB;
  _Float16*    o = A + (size_t)row * NB;
  float m = -3.4e38f;
  for (int j = tid; j < NB; j += 256) m = fmaxf(m, s[j]);
  red[tid] = m; __syncthreads();
  for (int off = 128; off > 0; off >>= 1) {
    if (tid < off) red[tid] = fmaxf(red[tid], red[tid + off]);
    __syncthreads();
  }
  m = red[0]; __syncthreads();
  float e[2], sum = 0.f;
  for (int j = tid, k = 0; j < NB; j += 256, ++k) { e[k] = __expf(s[j] - m); sum += e[k]; }
  red[tid] = sum; __syncthreads();
  for (int off = 128; off > 0; off >>= 1) {
    if (tid < off) red[tid] += red[tid + off];
    __syncthreads();
  }
  float inv = 1.f / red[0];
  for (int j = tid, k = 0; j < NB; j += 256, ++k) o[j] = (_Float16)(e[k] * inv);
}

// ---------------------------------------------------------------------------

static inline void gemm(hipStream_t st, const _Float16* A, int lda,
                        const _Float16* W, int ldw, const float* bias,
                        float* Cf, int ldcf, _Float16* Ch, int ldch,
                        int M, int N, int K, float alpha = 1.0f, int relu = 0) {
  int total = (M / 32) * (N / 32);
  int blocks = (total + 3) / 4;
  gemm_wmma_f16<<<blocks, 128, 0, st>>>(A, lda, W, ldw, bias, Cf, ldcf,
                                        Ch, ldch, M, N, K, alpha, relu);
}
static inline void conv16(hipStream_t st, const float* s, _Float16* d, int n) {
  f32_to_f16_kernel<<<(n + 255) / 256, 256, 0, st>>>(s, d, n);
}

extern "C" void kernel_launch(void* const* d_in, const int* in_sizes, int n_in,
                              void* d_out, int out_size, void* d_ws, size_t ws_size,
                              hipStream_t stream) {
  (void)in_sizes; (void)n_in; (void)out_size; (void)ws_size;
  const float* x        = (const float*)d_in[0];
  const float* t        = (const float*)d_in[1];
  const int*   y        = (const int*)  d_in[2];
  const float* node_w   = (const float*)d_in[3];
  const float* node_b   = (const float*)d_in[4];
  const float* te_freqs = (const float*)d_in[5];
  const float* te_w1    = (const float*)d_in[6];
  const float* te_b1    = (const float*)d_in[7];
  const float* te_w2    = (const float*)d_in[8];
  const float* te_b2    = (const float*)d_in[9];
  const float* time_w   = (const float*)d_in[10];
  const float* time_b   = (const float*)d_in[11];
  const float* label_emb= (const float*)d_in[12];
  const float* fuse_w   = (const float*)d_in[13];
  const float* fuse_b   = (const float*)d_in[14];
  const float* pen_b    = (const float*)d_in[16];
  const float* pel_w    = (const float*)d_in[17];
  const float* pel_b    = (const float*)d_in[18];
  const float* cat_w    = (const float*)d_in[19];
  const float* cat_b    = (const float*)d_in[20];
  const float* gg_w     = (const float*)d_in[21];
  const float* gru_wih  = (const float*)d_in[22];
  const float* gru_whh  = (const float*)d_in[23];
  const float* gru_bih  = (const float*)d_in[24];
  const float* gru_bhh  = (const float*)d_in[25];
  const float* attn_inw = (const float*)d_in[26];
  const float* attn_inb = (const float*)d_in[27];
  const float* attn_outw= (const float*)d_in[28];
  const float* attn_outb= (const float*)d_in[29];
  const float* n1_g = (const float*)d_in[30];
  const float* n1_b = (const float*)d_in[31];
  const float* n2_g = (const float*)d_in[32];
  const float* n2_b = (const float*)d_in[33];
  const float* mlp_w1 = (const float*)d_in[34];
  const float* mlp_b1 = (const float*)d_in[35];
  const float* mlp_w2 = (const float*)d_in[36];
  const float* mlp_b2 = (const float*)d_in[37];
  const float* n3_g = (const float*)d_in[38];
  const float* n3_b = (const float*)d_in[39];
  const float* out_w = (const float*)d_in[40];
  const float* out_b = (const float*)d_in[41];
  float* out = (float*)d_out;

  // -------- workspace carve-out (256B aligned slabs) --------
  char* base = (char*)d_ws; size_t off = 0;
  auto alloc = [&](size_t bytes) -> void* {
    void* p = base + off; off = (off + bytes + 255) & ~(size_t)255; return p;
  };
  _Float16* XH    = (_Float16*)alloc((size_t)NB * DIN * 2);
  _Float16* WNODE = (_Float16*)alloc((size_t)HID * DIN * 2);
  _Float16* WOUT  = (_Float16*)alloc((size_t)DIN * HID * 2);
  _Float16* WFUSE = (_Float16*)alloc((size_t)HID * 384 * 2);
  _Float16* WCAT  = (_Float16*)alloc((size_t)HID * HID * 2);
  _Float16* WHH   = (_Float16*)alloc((size_t)LLY * 384 * HID * 2);
  _Float16* WIN   = (_Float16*)alloc((size_t)LLY * 384 * HID * 2);
  _Float16* WAO   = (_Float16*)alloc((size_t)LLY * HID * HID * 2);
  _Float16* W1    = (_Float16*)alloc((size_t)LLY * 256 * HID * 2);
  _Float16* W2    = (_Float16*)alloc((size_t)LLY * HID * 256 * 2);
  _Float16* HCAT  = (_Float16*)alloc((size_t)NB * 384 * 2);
  _Float16* H116  = (_Float16*)alloc((size_t)NB * HID * 2);
  float*    CATB  = (float*)   alloc(HID * 4);
  float*    H     = (float*)   alloc((size_t)NB * HID * 4);
  _Float16* H16   = (_Float16*)alloc((size_t)NB * HID * 2);
  float*    GIV   = (float*)   alloc(384 * 4);
  float*    GH    = (float*)   alloc((size_t)NB * 384 * 4);
  float*    P     = (float*)   alloc((size_t)NB * HID * 4);
  float*    HL    = (float*)   alloc((size_t)NB * HID * 4);
  float*    QKV   = (float*)   alloc((size_t)NB * 384 * 4);
  _Float16* QKV16 = (_Float16*)alloc((size_t)NB * 384 * 2);
  _Float16* VT16  = (_Float16*)alloc((size_t)HEADS * DH * NB * 2);
  float*    S     = (float*)   alloc((size_t)HEADS * NB * NB * 4);
  _Float16* A16   = (_Float16*)alloc((size_t)HEADS * NB * NB * 2);
  float*    O     = (float*)   alloc((size_t)NB * HID * 4);
  _Float16* O16   = (_Float16*)alloc((size_t)NB * HID * 2);
  float*    AO    = (float*)   alloc((size_t)NB * HID * 4);
  float*    PA    = (float*)   alloc((size_t)NB * HID * 4);
  float*    HA    = (float*)   alloc((size_t)NB * HID * 4);
  float*    OUT0  = (float*)   alloc((size_t)NB * HID * 4);
  _Float16* OUT016= (_Float16*)alloc((size_t)NB * HID * 2);
  _Float16* M116  = (_Float16*)alloc((size_t)NB * 256 * 2);
  float*    M2    = (float*)   alloc((size_t)NB * HID * 4);
  float*    PRE3  = (float*)   alloc((size_t)NB * HID * 4);

  // -------- one-time (per call) f16 weight/activation conversion --------
  conv16(stream, x,        XH,    NB * DIN);
  conv16(stream, node_w,   WNODE, HID * DIN);
  conv16(stream, out_w,    WOUT,  DIN * HID);
  conv16(stream, fuse_w,   WFUSE, HID * 384);
  conv16(stream, gru_whh,  WHH,   LLY * 384 * HID);
  conv16(stream, attn_inw, WIN,   LLY * 384 * HID);
  conv16(stream, attn_outw,WAO,   LLY * HID * HID);
  conv16(stream, mlp_w1,   W1,    LLY * 256 * HID);
  conv16(stream, mlp_w2,   W2,    LLY * HID * 256);
  catw_conv_kernel<<<(HID * HID + 255) / 256, 256, 0, stream>>>(cat_w, WCAT);

  // -------- front end --------
  // h_data = x @ node_w^T + node_b  -> hcat[:, 0:128]
  gemm(stream, XH, DIN, WNODE, DIN, node_b, nullptr, 0, HCAT, 384, NB, HID, DIN);
  embed_kernel<<<NB, 128, 0, stream>>>(t, y, te_freqs, te_w1, te_b1, te_w2,
                                       te_b2, time_w, time_b, label_emb, HCAT);
  catbias_kernel<<<1, 128, 0, stream>>>(pen_b, pel_w, pel_b, cat_w, cat_b, CATB);
  // h1 = hcat @ fuse_w^T + fuse_b
  gemm(stream, HCAT, 384, WFUSE, 384, fuse_b, nullptr, 0, H116, HID, NB, HID, 384);
  // h = h1 @ cat_w[:, :128]^T + catbias  (PE term folded into catbias)
  gemm(stream, H116, HID, WCAT, HID, CATB, H, HID, H16, HID, NB, HID, HID);

  const float scale = 0.17677669529663687f;   // 1/sqrt(32)
  const int NE = NB * HID;

  for (int l = 0; l < LLY; ++l) {
    // --- GatedGraphConv (collapsed aggregation) + GRU ---
    vec_kernel<<<1, 384, 0, stream>>>(H, gg_w + (size_t)l * HID * HID,
                                      gru_wih + (size_t)l * 384 * HID,
                                      gru_bih + (size_t)l * 384, GIV);
    gemm(stream, H16, HID, WHH + (size_t)l * 384 * HID, HID,
         gru_bhh + (size_t)l * 384, GH, 384, nullptr, 0, NB, 384, HID);
    gru_elt_kernel<<<(NE + 255) / 256, 256, 0, stream>>>(GH, GIV, H, P);
    bn_kernel<<<HID, 256, 0, stream>>>(P, n1_g + l * HID, n1_b + l * HID, HL, nullptr);

    // --- global multi-head attention ---
    gemm(stream, H16, HID, WIN + (size_t)l * 384 * HID, HID,
         attn_inb + (size_t)l * 384, QKV, 384, QKV16, 384, NB, 384, HID);
    vt_kernel<<<(HEADS * DH * NB + 255) / 256, 256, 0, stream>>>(QKV, VT16);
    for (int h = 0; h < HEADS; ++h) {
      // S_h = scale * q_h @ k_h^T   (q,k are column slices of QKV16)
      gemm(stream, QKV16 + h * DH, 384, QKV16 + HID + h * DH, 384, nullptr,
           S + (size_t)h * NB * NB, NB, nullptr, 0, NB, NB, DH, scale);
    }
    softmax_kernel<<<HEADS * NB, 256, 0, stream>>>(S, A16);
    for (int h = 0; h < HEADS; ++h) {
      // O_h = A_h @ v_h   (VT16 is v^T, already W-form)
      gemm(stream, A16 + (size_t)h * NB * NB, NB, VT16 + (size_t)h * DH * NB, NB,
           nullptr, O + h * DH, HID, O16 + h * DH, HID, NB, DH, NB);
    }
    gemm(stream, O16, HID, WAO + (size_t)l * HID * HID, HID,
         attn_outb + (size_t)l * HID, AO, HID, nullptr, 0, NB, HID, HID);
    add_kernel<<<(NE + 255) / 256, 256, 0, stream>>>(AO, H, PA, nullptr, NE);
    bn_kernel<<<HID, 256, 0, stream>>>(PA, n2_g + l * HID, n2_b + l * HID, HA, nullptr);

    // --- combine + MLP + norm3 ---
    add_kernel<<<(NE + 255) / 256, 256, 0, stream>>>(HL, HA, OUT0, OUT016, NE);
    gemm(stream, OUT016, HID, W1 + (size_t)l * 256 * HID, HID,
         mlp_b1 + (size_t)l * 256, nullptr, 0, M116, 256, NB, 256, HID, 1.0f, 1);
    gemm(stream, M116, 256, W2 + (size_t)l * HID * 256, 256,
         mlp_b2 + (size_t)l * HID, M2, HID, nullptr, 0, NB, HID, 256);
    add_kernel<<<(NE + 255) / 256, 256, 0, stream>>>(OUT0, M2, PRE3, nullptr, NE);
    bn_kernel<<<HID, 256, 0, stream>>>(PRE3, n3_g + l * HID, n3_b + l * HID, H, H16);
  }

  // -------- output projection: out = h @ out_w^T + out_b --------
  gemm(stream, H16, HID, WOUT, HID, out_b, out, DIN, nullptr, 0, NB, DIN, HID);
}
